// CountingDecoder_13013750906946
// MI455X (gfx1250) — matrix operand order repaired and checked by reading, not compiled, not run on hardware
//
#include <hip/hip_runtime.h>
#include <math.h>

// ---------------------------------------------------------------------------
// CountingDecoder on MI455X (gfx1250, wave32).
// Heavy op: dynamic MoE conv as per-sample implicit GEMM on
// v_wmma_f32_16x16x32_f16. Weights are pre-combined per sample to f16 so the
// GEMM A-stage is pure b128 copies (adaptive on ws_size).
// ---------------------------------------------------------------------------

typedef __attribute__((ext_vector_type(16))) _Float16 v16h;
typedef __attribute__((ext_vector_type(8)))  _Float16 v8h;
typedef __attribute__((ext_vector_type(8)))  float    v8f;

#define NEXP   4
#define INCH   128
#define MIDCH  512
#define OUTCH  111
#define HW     32
#define PIX    1024
#define NB     16

// flattened GEMM K dimension: 128*(9+25+49)
#define KTOT   10624
#define KB3    0
#define KB5    1152
#define KB7    4352

// Conv LDS row strides (halves): 72*2 = 144 B -> all fragment sub-loads stay
// 16-byte aligned (ds_load_b128) and the stride is not a power of two.
#define ASTR 72
#define BSTR 72
// Pred kernel uses 32-K chunks with stride 40 (80 B).
#define PSTR 40

// ---- WMMA fragment loaders (contiguous k order, same for A and B) -----------------
// A 16x32 f16: lanes 0-15 -> K0..15 of row M; lanes 16-31 -> K16..31
// B 32x16 f16: lanes 0-15 -> K0..15 of col N; lanes 16-31 -> K16..31
__device__ __forceinline__ v16h frag_join(const v8h* lo, const v8h* hi) {
  return __builtin_shufflevector(*lo, *hi,
      0,1,2,3,4,5,6,7,8,9,10,11,12,13,14,15);
}
__device__ __forceinline__ v16h load_frag_A(const _Float16* row, int lane) {
  const int koff = (lane & 16) ? 16 : 0;
  return frag_join((const v8h*)(row + koff), (const v8h*)(row + koff + 8));
}
__device__ __forceinline__ v16h load_frag_B(const _Float16* col, int lane) {
  const int koff = (lane & 16) ? 16 : 0;
  return frag_join((const v8h*)(col + koff), (const v8h*)(col + koff + 8));
}

// ---------------------------------------------------------------------------
// Kernel 1: router — pooled mean + softmax over 4 experts. One block / sample.
// ---------------------------------------------------------------------------
__global__ __launch_bounds__(128)
void router_kernel(const float* __restrict__ x,
                   const float* __restrict__ rw,
                   const float* __restrict__ rb,
                   float* __restrict__ routing) {
  const int b = blockIdx.x;
  const int c = threadIdx.x;              // 128 threads == 128 channels
  __shared__ float pooled[INCH];
  __shared__ float logits[NEXP];
  const float* row = x + ((size_t)b * INCH + c) * PIX;
  float s = 0.f;
  for (int p = 0; p < PIX; ++p) s += row[p];
  pooled[c] = s * (1.0f / PIX);
  __syncthreads();
  if (c < NEXP) {
    float l = rb[c];
    for (int j = 0; j < INCH; ++j) l += pooled[j] * rw[c * INCH + j];
    logits[c] = l;
  }
  __syncthreads();
  if (c == 0) {
    float mx = logits[0];
    for (int e = 1; e < NEXP; ++e) mx = fmaxf(mx, logits[e]);
    float es[NEXP], sum = 0.f;
    for (int e = 0; e < NEXP; ++e) { es[e] = __expf(logits[e] - mx); sum += es[e]; }
    for (int e = 0; e < NEXP; ++e) routing[b * NEXP + e] = es[e] / sum;
  }
}

// ---------------------------------------------------------------------------
// Kernel 2a: pre-combine expert weights per sample into f16:
//   wcomb[b, o, k] = sum_e routing[b,e] * W_e[o, k],  k in flattened 3/5/7 order
// Each thread reads the 4 expert values once and emits all 16 samples.
// ---------------------------------------------------------------------------
__global__ __launch_bounds__(256)
void combine_weights_kernel(const float* __restrict__ wa,   // 3x3 bank
                            const float* __restrict__ wb,   // 5x5 bank
                            const float* __restrict__ wc,   // 7x7 bank
                            const float* __restrict__ routing,
                            _Float16* __restrict__ wcomb) {
  const int o = blockIdx.y;
  const int k = blockIdx.x * 256 + threadIdx.x;
  __shared__ float r[NB * NEXP];
  if (threadIdx.x < NB * NEXP) r[threadIdx.x] = routing[threadIdx.x];
  __syncthreads();
  if (k >= KTOT) return;
  const float* W; int ks2; int kl;
  if (k < KB5)      { W = wa; ks2 = 9;  kl = k; }
  else if (k < KB7) { W = wb; ks2 = 25; kl = k - KB5; }
  else              { W = wc; ks2 = 49; kl = k - KB7; }
  const size_t es   = (size_t)MIDCH * INCH * ks2;       // expert stride
  const size_t base = (size_t)o * (INCH * ks2) + kl;
  const float e0 = W[base];
  const float e1 = W[base + es];
  const float e2 = W[base + 2 * es];
  const float e3 = W[base + 3 * es];
  #pragma unroll
  for (int b = 0; b < NB; ++b) {
    const float v = r[b*4+0]*e0 + r[b*4+1]*e1 + r[b*4+2]*e2 + r[b*4+3]*e3;
    wcomb[((size_t)(b * MIDCH + o)) * KTOT + k] = (_Float16)v;
  }
}

// Kernel 2b: f32 -> f16 copy of x (kills per-chunk converts in the B-stage).
__global__ __launch_bounds__(256)
void f32_to_f16_kernel(const float* __restrict__ src,
                       _Float16* __restrict__ dst, int n) {
  const int i = blockIdx.x * 256 + threadIdx.x;
  if (i < n) dst[i] = (_Float16)src[i];
}

// ---------------------------------------------------------------------------
// Kernel 3: dynamic conv as implicit GEMM, K-step 64 (two WMMA sub-chunks per
// barrier interval). PRE=true: A-stage = b128 copies from wcomb, B-stage from
// f16 x. PRE=false: on-the-fly combine fallback (small-workspace path).
// ---------------------------------------------------------------------------
template <int S, bool PRE>
__device__ void conv_accumulate(const float* __restrict__ x,
                                const _Float16* __restrict__ xh,
                                const float* __restrict__ W,      // raw bank
                                const _Float16* __restrict__ Wc,  // wcomb + b + kbase
                                float r0, float r1, float r2, float r3,
                                int b, int Obase, int Pbase,
                                _Float16* Alds, _Float16* Blds,
                                v8f (&acc)[4][4]) {
  constexpr int KS2 = S * S;
  constexpr int K   = INCH * KS2;         // 1152/3200/6272: all %64 == 0
  constexpr int PAD = S / 2;
  const int tid   = threadIdx.x;
  const int lane  = tid & 31;
  const int wid   = tid >> 5;
  const int waveM = wid & 1;
  const int waveN = wid >> 1;

  const int nl = tid;                     // this thread's pixel column (B tile)
  const int p  = Pbase + nl;
  const int py = p >> 5, px = p & 31;

  for (int k0 = 0; k0 < K; k0 += 64) {
    // ---- Stage A: [128 o x 64 k] f16 ----
    if (PRE) {
      // pure vector copy: one v8h global load + one v8h LDS store per step
      #pragma unroll
      for (int t = tid; t < 128 * 8; t += 256) {
        const int ol = t >> 3;
        const int kc = (t & 7) * 8;
        const v8h v = *(const v8h*)(Wc + (size_t)(Obase + ol) * KTOT + k0 + kc);
        *(v8h*)(Alds + ol * ASTR + kc) = v;
      }
    } else {
      const size_t estride = (size_t)MIDCH * INCH * KS2;
      for (int idx = tid; idx < 128 * 64; idx += 256) {
        const int ol = idx >> 6;
        const int kk = idx & 63;
        const size_t base = (size_t)(Obase + ol) * (INCH * KS2) + (k0 + kk);
        const float wv = r0 * W[base]
                       + r1 * W[base +     estride]
                       + r2 * W[base + 2 * estride]
                       + r3 * W[base + 3 * estride];
        Alds[ol * ASTR + kk] = (_Float16)wv;
      }
    }
    // ---- Stage B: implicit im2col [256 n x 64 k] f16 (n-major) ----
    #pragma unroll
    for (int kk = 0; kk < 64; ++kk) {
      const int k  = k0 + kk;
      const int i  = k / KS2;             // compile-time divisors
      const int rr = k - i * KS2;
      const int dy = rr / S;
      const int dx = rr - dy * S;
      const int sy = py + dy - PAD;
      const int sx = px + dx - PAD;
      const bool in = ((unsigned)sy < (unsigned)HW) & ((unsigned)sx < (unsigned)HW);
      _Float16 v = (_Float16)0.0f;
      if (PRE) {
        if (in) v = xh[((size_t)b * INCH + i) * PIX + (sy << 5) + sx];
      } else {
        if (in) v = (_Float16)x[((size_t)b * INCH + i) * PIX + (sy << 5) + sx];
      }
      Blds[nl * BSTR + kk] = v;
    }
    __syncthreads();
    #pragma unroll
    for (int kc = 0; kc < 64; kc += 32) {
      v16h afrag[4];
      #pragma unroll
      for (int mi = 0; mi < 4; ++mi)
        afrag[mi] = load_frag_A(&Alds[(waveM*64 + mi*16 + (lane & 15)) * ASTR + kc], lane);
      #pragma unroll
      for (int ni = 0; ni < 4; ++ni) {
        const v16h bfrag = load_frag_B(&Blds[(waveN*64 + ni*16 + (lane & 15)) * BSTR + kc], lane);
        #pragma unroll
        for (int mi = 0; mi < 4; ++mi)
          acc[mi][ni] = __builtin_amdgcn_wmma_f32_16x16x32_f16(
              false, afrag[mi], false, bfrag, (short)0, acc[mi][ni], false, false);
      }
    }
    __syncthreads();
  }
}

template <bool PRE>
__global__ __launch_bounds__(256)
void dynconv_wmma_kernel(const float* __restrict__ x,
                         const _Float16* __restrict__ xh,
                         const float* __restrict__ w3,
                         const float* __restrict__ w5,
                         const float* __restrict__ w7,
                         const _Float16* __restrict__ wcomb,
                         const float* __restrict__ routing,
                         _Float16* __restrict__ yh) {
  __shared__ alignas(16) _Float16 Alds[128 * ASTR];
  __shared__ alignas(16) _Float16 Blds[256 * BSTR];
  const int b     = blockIdx.z;
  const int Obase = blockIdx.y * 128;
  const int Pbase = blockIdx.x * 256;
  const int tid   = threadIdx.x;
  const int lane  = tid & 31;
  const int wid   = tid >> 5;
  const int waveM = wid & 1;
  const int waveN = wid >> 1;

  float r0 = 0.f, r1 = 0.f, r2 = 0.f, r3 = 0.f;
  if (!PRE) {
    r0 = routing[b * NEXP + 0];
    r1 = routing[b * NEXP + 1];
    r2 = routing[b * NEXP + 2];
    r3 = routing[b * NEXP + 3];
  }
  const _Float16* Wcb = wcomb + (size_t)b * MIDCH * KTOT;

  v8f zero;
  #pragma unroll
  for (int i = 0; i < 8; ++i) zero[i] = 0.0f;
  v8f acc[4][4];
  #pragma unroll
  for (int mi = 0; mi < 4; ++mi)
    #pragma unroll
    for (int ni = 0; ni < 4; ++ni) acc[mi][ni] = zero;

  conv_accumulate<3, PRE>(x, xh, w3, Wcb + KB3, r0, r1, r2, r3, b, Obase, Pbase, Alds, Blds, acc);
  conv_accumulate<5, PRE>(x, xh, w5, Wcb + KB5, r0, r1, r2, r3, b, Obase, Pbase, Alds, Blds, acc);
  conv_accumulate<7, PRE>(x, xh, w7, Wcb + KB7, r0, r1, r2, r3, b, Obase, Pbase, Alds, Blds, acc);

  // Epilogue: D layout — lane&15 = N, VGPR r = M row (+8 for lanes 16-31).
  #pragma unroll
  for (int mi = 0; mi < 4; ++mi)
    #pragma unroll
    for (int ni = 0; ni < 4; ++ni)
      #pragma unroll
      for (int r = 0; r < 8; ++r) {
        const int o = Obase + waveM * 64 + mi * 16 + r + ((lane & 16) ? 8 : 0);
        const int p = Pbase + waveN * 64 + ni * 16 + (lane & 15);
        yh[((size_t)b * MIDCH + o) * PIX + p] = (_Float16)acc[mi][ni][r];
      }
}

// ---------------------------------------------------------------------------
// Kernel 4: per-channel spatial mean of y (input to channel attention).
// ---------------------------------------------------------------------------
__global__ __launch_bounds__(256)
void ca_reduce_kernel(const _Float16* __restrict__ yh,
                      float* __restrict__ ca_mean) {
  const int o = blockIdx.x;
  const int b = blockIdx.y;
  __shared__ float sdata[256];
  const _Float16* row = yh + ((size_t)b * MIDCH + o) * PIX;
  float s = 0.f;
  for (int p = threadIdx.x; p < PIX; p += 256) s += (float)row[p];
  sdata[threadIdx.x] = s;
  __syncthreads();
  for (int st = 128; st > 0; st >>= 1) {
    if (threadIdx.x < st) sdata[threadIdx.x] += sdata[threadIdx.x + st];
    __syncthreads();
  }
  if (threadIdx.x == 0) ca_mean[b * MIDCH + o] = sdata[0] * (1.0f / PIX);
}

// ---------------------------------------------------------------------------
// Kernel 5: CA MLP (512->32 relu ->512 sigmoid) + mask projection per pixel.
// ---------------------------------------------------------------------------
__global__ __launch_bounds__(256)
void ca_mask_kernel(const float* __restrict__ ca_mean,
                    const float* __restrict__ fc1w, const float* __restrict__ fc1b,
                    const float* __restrict__ fc2w, const float* __restrict__ fc2b,
                    const float* __restrict__ mask,
                    const float* __restrict__ mpw,  const float* __restrict__ mpb,
                    float* __restrict__ ca, float* __restrict__ mask_feat) {
  const int b = blockIdx.x;
  const int t = threadIdx.x;
  __shared__ float cm[MIDCH];
  __shared__ float h[32];
  for (int i = t; i < MIDCH; i += 256) cm[i] = ca_mean[b * MIDCH + i];
  __syncthreads();
  if (t < 32) {
    float a = fc1b[t];
    for (int j = 0; j < MIDCH; ++j) a += fc1w[t * MIDCH + j] * cm[j];
    h[t] = fmaxf(a, 0.0f);
  }
  __syncthreads();
  for (int o = t; o < MIDCH; o += 256) {
    float a = fc2b[o];
    for (int j = 0; j < 32; ++j) a += fc2w[o * 32 + j] * h[j];
    ca[b * MIDCH + o] = 1.0f / (1.0f + __expf(-a));
  }
  for (int p = t; p < PIX; p += 256) {
    float a = mpb[0];
    for (int c2 = 0; c2 < INCH; ++c2)
      a += mask[((size_t)b * INCH + c2) * PIX + p] * mpw[c2];
    mask_feat[b * PIX + p] = a;
  }
}

// ---------------------------------------------------------------------------
// Kernel 6: zero the count region (harness poisons d_out only once).
// ---------------------------------------------------------------------------
__global__ __launch_bounds__(256)
void zero_count_kernel(float* __restrict__ cnt, int n) {
  const int i = blockIdx.x * 256 + threadIdx.x;
  if (i < n) cnt[i] = 0.0f;
}

// ---------------------------------------------------------------------------
// Kernel 7: pred layer as WMMA GEMM with ca folded into A and mask_feat in
// the epilogue:  d = sigmoid(mask_feat * ((pred_w*ca) x y));  count = Σ_p d.
// ---------------------------------------------------------------------------
__global__ __launch_bounds__(256)
void pred_wmma_kernel(const _Float16* __restrict__ yh,
                      const float* __restrict__ pred_w,
                      const float* __restrict__ ca,
                      const float* __restrict__ mask_feat,
                      float* __restrict__ count,
                      float* __restrict__ dmap) {
  __shared__ alignas(16) _Float16 Alds[128 * PSTR];
  __shared__ alignas(16) _Float16 Blds[256 * PSTR];
  const int b     = blockIdx.y;
  const int Pbase = blockIdx.x * 256;
  const int tid   = threadIdx.x;
  const int lane  = tid & 31;
  const int wid   = tid >> 5;
  const int waveM = wid & 1;
  const int waveN = wid >> 1;

  v8f zero;
  #pragma unroll
  for (int i = 0; i < 8; ++i) zero[i] = 0.0f;
  v8f acc[4][4];
  #pragma unroll
  for (int mi = 0; mi < 4; ++mi)
    #pragma unroll
    for (int ni = 0; ni < 4; ++ni) acc[mi][ni] = zero;

  for (int k0 = 0; k0 < MIDCH; k0 += 32) {
    for (int idx = tid; idx < 128 * 32; idx += 256) {
      const int ol = idx >> 5;
      const int kk = idx & 31;
      const int c  = k0 + kk;
      float v = 0.0f;
      if (ol < OUTCH) v = pred_w[ol * MIDCH + c] * ca[b * MIDCH + c];
      Alds[ol * PSTR + kk] = (_Float16)v;
    }
    {
      const int nl = tid;
      const _Float16* src = yh + ((size_t)b * MIDCH + k0) * PIX + Pbase + nl;
      #pragma unroll
      for (int kk = 0; kk < 32; ++kk)
        Blds[nl * PSTR + kk] = src[(size_t)kk * PIX];
    }
    __syncthreads();
    v16h afrag[4];
    #pragma unroll
    for (int mi = 0; mi < 4; ++mi)
      afrag[mi] = load_frag_A(&Alds[(waveM*64 + mi*16 + (lane & 15)) * PSTR], lane);
    #pragma unroll
    for (int ni = 0; ni < 4; ++ni) {
      const v16h bfrag = load_frag_B(&Blds[(waveN*64 + ni*16 + (lane & 15)) * PSTR], lane);
      #pragma unroll
      for (int mi = 0; mi < 4; ++mi)
        acc[mi][ni] = __builtin_amdgcn_wmma_f32_16x16x32_f16(
            false, afrag[mi], false, bfrag, (short)0, acc[mi][ni], false, false);
    }
    __syncthreads();
  }

  #pragma unroll
  for (int mi = 0; mi < 4; ++mi) {
    #pragma unroll
    for (int r = 0; r < 8; ++r) {
      const int o = waveM * 64 + mi * 16 + r + ((lane & 16) ? 8 : 0);
      const bool valid = (o < OUTCH);
      float s = 0.0f;
      #pragma unroll
      for (int ni = 0; ni < 4; ++ni) {
        const int p   = Pbase + waveN * 64 + ni * 16 + (lane & 15);
        const float g = acc[mi][ni][r];
        const float mf = mask_feat[b * PIX + p];
        const float dv = 1.0f / (1.0f + __expf(-g * mf));
        if (valid) {
          dmap[((size_t)(b * OUTCH + o)) * PIX + p] = dv;
          s += dv;
        }
      }
      s += __shfl_xor(s, 1, 32);
      s += __shfl_xor(s, 2, 32);
      s += __shfl_xor(s, 4, 32);
      s += __shfl_xor(s, 8, 32);
      if (valid && ((lane & 15) == 0))
        atomicAdd(&count[b * OUTCH + o], s);
    }
  }
}

// ---------------------------------------------------------------------------
// Host-side launcher. Workspace layout (byte offsets, all 16B aligned):
//   routing      @ 0          (256 B)
//   ca_mean      @ 256        (32 KB)
//   ca           @ 33024      (32 KB)
//   mask_feat    @ 65792      (64 KB)
//   yh   (f16)   @ 131328     (16.78 MB)
//   xh   (f16)   @ 16908544   (4.19 MB)      [PRE path only]
//   wcomb(f16)   @ 21102848   (174.06 MB)    [PRE path only]
// ---------------------------------------------------------------------------
extern "C" void kernel_launch(void* const* d_in, const int* in_sizes, int n_in,
                              void* d_out, int out_size, void* d_ws, size_t ws_size,
                              hipStream_t stream) {
  const float* x     = (const float*)d_in[0];
  const float* mask  = (const float*)d_in[1];
  const float* w3    = (const float*)d_in[2];
  const float* w5    = (const float*)d_in[3];
  const float* w7    = (const float*)d_in[4];
  const float* rw    = (const float*)d_in[5];
  const float* rb    = (const float*)d_in[6];
  const float* fc1w  = (const float*)d_in[7];
  const float* fc1b  = (const float*)d_in[8];
  const float* fc2w  = (const float*)d_in[9];
  const float* fc2b  = (const float*)d_in[10];
  const float* mpw   = (const float*)d_in[11];
  const float* mpb   = (const float*)d_in[12];
  const float* predw = (const float*)d_in[13];

  float* out   = (float*)d_out;
  float* count = out;                 // [16,111]
  float* dmap  = out + NB * OUTCH;    // [16,111,32,32]

  char* ws = (char*)d_ws;
  float*    routing   = (float*)(ws + 0);
  float*    ca_mean   = (float*)(ws + 256);
  float*    ca        = (float*)(ws + 33024);
  float*    mask_feat = (float*)(ws + 65792);
  _Float16* yh        = (_Float16*)(ws + 131328);
  _Float16* xh        = (_Float16*)(ws + 16908544ull);
  _Float16* wcomb     = (_Float16*)(ws + 21102848ull);
  const bool pre = (ws_size >= 195166464ull);   // room for xh + wcomb?

  router_kernel<<<dim3(NB), dim3(128), 0, stream>>>(x, rw, rb, routing);

  if (pre) {
    combine_weights_kernel<<<dim3((KTOT + 255) / 256, MIDCH), dim3(256), 0, stream>>>(
        w3, w5, w7, routing, wcomb);
    f32_to_f16_kernel<<<dim3((NB * INCH * PIX) / 256), dim3(256), 0, stream>>>(
        x, xh, NB * INCH * PIX);
    dynconv_wmma_kernel<true><<<dim3(4, 4, NB), dim3(256), 0, stream>>>(
        x, xh, w3, w5, w7, wcomb, routing, yh);
  } else {
    dynconv_wmma_kernel<false><<<dim3(4, 4, NB), dim3(256), 0, stream>>>(
        x, xh, w3, w5, w7, wcomb, routing, yh);
  }

  ca_reduce_kernel<<<dim3(MIDCH, NB), dim3(256), 0, stream>>>(yh, ca_mean);

  ca_mask_kernel<<<dim3(NB), dim3(256), 0, stream>>>(
      ca_mean, fc1w, fc1b, fc2w, fc2b, mask, mpw, mpb, ca, mask_feat);

  zero_count_kernel<<<dim3((NB * OUTCH + 255) / 256), dim3(256), 0, stream>>>(
      count, NB * OUTCH);

  pred_wmma_kernel<<<dim3(4, NB), dim3(256), 0, stream>>>(
      yh, predw, ca, mask_feat, count, dmap);

  (void)in_sizes; (void)n_in; (void)out_size; (void)ws_size;
}
